// BaseHashCode_72756745994542
// MI455X (gfx1250) — 8-bit matrix operand order repaired and checked by reading, compiled, not run on hardware
//
#include <hip/hip_runtime.h>
#include <hip/hip_bf16.h>

typedef int   v8i __attribute__((ext_vector_type(8)));
typedef int   v4i __attribute__((ext_vector_type(4)));
typedef float v4f __attribute__((ext_vector_type(4)));

#define PRIME_U      1000003u
#define SEQ_LEN      64
#define WAVES_PER_WG 8
#define WG_SIZE      256
#define LDS_STRIDE   65
#define LDS_PER_WAVE (16 * LDS_STRIDE + 16)

__global__ __launch_bounds__(WG_SIZE) void BaseHashCode_kernel(
    const int* __restrict__ seq, const int* __restrict__ acoef,
    const int* __restrict__ bptr, float* __restrict__ out, int n_blocks)
{
    __shared__ unsigned s_lds[WAVES_PER_WG * LDS_PER_WAVE];

    const int lane = threadIdx.x & 31;
    const int wav  = threadIdx.x >> 5;
    unsigned* s_h   = &s_lds[wav * LDS_PER_WAVE];
    unsigned* s_len = s_h + 16 * LDS_STRIDE;

    const int h16 = lane >> 4;
    const int l16 = lane & 15;
    const unsigned badd = (unsigned)bptr[0];

    v8i Cb;
    #pragma unroll
    for (int r = 0; r < 8; ++r) Cb[r] = (int)badd;

    v8i B[3][4];
    {
        int aK[8][4];
        #pragma unroll
        for (int j = 0; j < 8; ++j) {
            const int K0 = ((j >> 2) * 32) + h16 * 16 + ((j & 3) * 4);
            v4i av = *(const v4i*)(acoef + K0);
            aK[j][0] = av.x; aK[j][1] = av.y; aK[j][2] = av.z; aK[j][3] = av.w;
        }
        #pragma unroll
        for (int p = 0; p < 3; ++p)
            #pragma unroll
            for (int c = 0; c < 4; ++c) {
                const int tcol = c * 16 + l16;
                #pragma unroll
                for (int j = 0; j < 8; ++j) {
                    const int K0 = ((j >> 2) * 32) + h16 * 16 + ((j & 3) * 4);
                    unsigned w = 0u;
                    #pragma unroll
                    for (int bb = 0; bb < 4; ++bb) {
                        const unsigned byte = ((unsigned)aK[j][bb] >> (8 * p)) & 0xFFu;
                        w |= ((K0 + bb) <= tcol ? byte : 0u) << (8 * bb);
                    }
                    B[p][c][j] = (int)w;
                }
            }
    }

    const int wave_id = blockIdx.x * WAVES_PER_WG + wav;
    const int n_waves = (int)gridDim.x * WAVES_PER_WG;

    for (int blk = wave_id; blk < n_blocks; blk += n_waves) {
        const int* rowp = seq + (size_t)blk * (16 * SEQ_LEN);

        const int nxt = blk + n_waves;
        if (nxt < n_blocks)
            __builtin_prefetch(seq + (size_t)nxt * (16 * SEQ_LEN) + lane * 32, 0, 0);

        v8i A;
        unsigned nzsum = 0u;
        #pragma unroll
        for (int j = 0; j < 8; ++j) {
            const int K0 = ((j >> 2) * 32) + h16 * 16 + ((j & 3) * 4);
            v4i xv = __builtin_nontemporal_load((const v4i*)(rowp + l16 * SEQ_LEN + K0));
            const unsigned t0 = __builtin_amdgcn_perm((unsigned)xv.y, (unsigned)xv.x, 0x00000400u);
            const unsigned t1 = __builtin_amdgcn_perm((unsigned)xv.w, (unsigned)xv.z, 0x00000400u);
            const unsigned w  = __builtin_amdgcn_perm(t1, t0, 0x05040100u);
            A[j] = (int)w;
            nzsum += (w | (w >> 1) | (w >> 2)) & 0x01010101u;
        }
        const int cnt = (int)((nzsum * 0x01010101u) >> 24);
        const int len = cnt + __shfl_xor(cnt, 16, 32);
        s_len[l16] = (unsigned)len;

        #pragma unroll
        for (int c = 0; c < 4; ++c) {
            v8i z = {};
            v8i d0 = __builtin_amdgcn_wmma_i32_16x16x64_iu8(false, A, false, B[0][c], Cb, false, false);
            v8i d1 = __builtin_amdgcn_wmma_i32_16x16x64_iu8(false, A, false, B[1][c], z,  false, false);
            v8i d2 = __builtin_amdgcn_wmma_i32_16x16x64_iu8(false, A, false, B[2][c], z,  false, false);
            #pragma unroll
            for (int r = 0; r < 8; ++r) {
                const unsigned acc = (unsigned)d0[r] + ((unsigned)d1[r] << 8) +
                                     ((unsigned)d2[r] << 16);
                const unsigned hsh = (acc % PRIME_U) & 0xFFFFu;
                const int m = r + 8 * h16;
                s_h[m * LDS_STRIDE + c * 16 + l16] = hsh;
            }
        }
        __builtin_amdgcn_wave_barrier();

        float* outp = out + (size_t)blk * (16 * SEQ_LEN);
        #pragma unroll
        for (int r = 0; r < 8; ++r) {
            const int m     = 2 * r + h16;
            const int t0    = l16 * 4;
            const int len_m = (int)s_len[m];
            v4f v;
            #pragma unroll
            for (int e = 0; e < 4; ++e) {
                const int t    = t0 + e;
                const int teff = (t < len_m) ? t : ((len_m - 1) & 63);
                v[e] = (float)s_h[m * LDS_STRIDE + teff];
            }
            __builtin_nontemporal_store(v, (v4f*)(outp + m * SEQ_LEN + t0));
        }
    }
}

extern "C" void kernel_launch(void* const* d_in, const int* in_sizes, int n_in,
                              void* d_out, int out_size, void* d_ws, size_t ws_size,
                              hipStream_t stream) {
    const int* seq = (const int*)d_in[0];
    const int* a   = (const int*)d_in[1];
    const int* b   = (const int*)d_in[2];
    float*     out = (float*)d_out;

    const int n_blocks = in_sizes[0] >> 10;
    int grid = (n_blocks + WAVES_PER_WG - 1) / WAVES_PER_WG;
    if (grid > 2048) grid = 2048;
    if (grid < 1)    grid = 1;

    BaseHashCode_kernel<<<dim3(grid), dim3(WG_SIZE), 0, stream>>>(seq, a, b, out, n_blocks);
}
